// GauntTensorProductS2Grid_58609123721876
// MI455X (gfx1250) — matrix operand order repaired and checked by reading, not compiled, not run on hardware
//
#include <hip/hip_runtime.h>
#include <hip/hip_bf16.h>

typedef _Float16 v16h __attribute__((ext_vector_type(16)));
typedef _Float16 v8h  __attribute__((ext_vector_type(8)));
typedef float    v8f  __attribute__((ext_vector_type(8)));

// Problem geometry (compile-time; reference: LMAX=8, RES_BETA=32, RES_ALPHA=63)
constexpr int NDIM = 81;   // (LMAX+1)^2
constexpr int KP   = 96;   // i-dim padded to 3 x 32 (WMMA K-step)
constexpr int RB   = 32;   // res_beta
constexpr int RA   = 63;   // res_alpha
constexpr int AP   = 64;   // alpha padded to 2 x 32
constexpr int WAVES = 2;   // waves per workgroup (wave32)
constexpr int TPB   = WAVES * 32;

union V16 { v16h v; v8h h[2]; };

// D = A(16xK f16) * B(Kx16 f16) + C(f32), K accumulated in steps of 32.
// Fragment layouts (wave32):
//   A: lane L -> row M = L&15; half h -> K = 16*(L>>4) + h
//   B: lane L -> col N = L&15; half h -> K = 16*(L>>4) + h
//   C: lane L -> col N = L&15; vgpr r -> row M = r + 8*(L>>4)
__global__ __launch_bounds__(TPB) void gaunt_s2_wmma(
    const float* __restrict__ x1, const float* __restrict__ x2,
    const float* __restrict__ shb, const float* __restrict__ sha,
    const float* __restrict__ qw, float* __restrict__ out, int batch)
{
  __shared__ __align__(16) _Float16 s_Bf[12 * 32 * 16];   // forward B frags: sha^T, f = ks*4 + nt
  __shared__ __align__(16) _Float16 s_Bb[12 * 32 * 16];   // backward B frags: sha,  f = ks*6 + nt
  __shared__ __align__(16) _Float16 s_shb[RB * KP];       // shb (f16, K-padded)
  __shared__ __align__(16) float    s_shbw[RB * KP];      // shb * qw (f32 output scale)
  __shared__ __align__(16) _Float16 s_x1[WAVES][16 * KP]; // per-wave x tiles (f16)
  __shared__ __align__(16) _Float16 s_x2[WAVES][16 * KP];
  __shared__ __align__(16) _Float16 s_g [WAVES][16 * AP]; // per-wave transpose scratch

  const int tid = threadIdx.x;

  // ---- per-WG basis tables (sha/shb tiny; stays L2-resident across WGs) ----
  for (int idx = tid; idx < RB * KP; idx += TPB) {
    int b = idx / KP, k = idx % KP;
    float v = (k < NDIM) ? shb[b * NDIM + k] : 0.f;
    s_shb[idx]  = (_Float16)v;
    s_shbw[idx] = v * qw[b];
  }
  for (int idx = tid; idx < 12 * 512; idx += TPB) {
    int f = idx >> 9, rem = idx & 511, l = rem >> 4, h = rem & 15;
    { // forward: B[k=i][n=a] = sha[a][i]
      int ks = f >> 2, nt = f & 3;
      int k = ks * 32 + ((l >> 4) << 4) + h;
      int a = nt * 16 + (l & 15);
      s_Bf[idx] = (_Float16)((a < RA && k < NDIM) ? sha[a * NDIM + k] : 0.f);
    }
    { // backward: B[k=a][n=i] = sha[a][i]
      int ks = f / 6, nt = f % 6;
      int a = ks * 32 + ((l >> 4) << 4) + h;
      int i = nt * 16 + (l & 15);
      s_Bb[idx] = (_Float16)((a < RA && i < NDIM) ? sha[a * NDIM + i] : 0.f);
    }
  }
  __syncthreads();

  const int wv   = tid >> 5;
  const int lane = tid & 31;
  const int hi   = lane >> 4;
  const int mrow = lane & 15;
  const int z0   = (blockIdx.x * WAVES + wv) * 16;
  if (z0 >= batch) return;  // wave-uniform: EXEC stays all-ones for WMMA

  // ---- stage this wave's 16-row x tiles into LDS (f16, zero K-pad) ----
  for (int idx = lane; idx < 16 * KP; idx += 32) {
    int mm = idx / KP, k = idx % KP;
    float a = 0.f, b = 0.f;
    if (k < NDIM) {
      size_t off = (size_t)(z0 + mm) * NDIM + k;
      a = x1[off];
      b = x2[off];
    }
    s_x1[wv][idx] = (_Float16)a;
    s_x2[wv][idx] = (_Float16)b;
  }
  asm volatile("" ::: "memory");  // keep DS stores ahead of DS reads (in-order in-wave)

  v8f acc[6] = {};  // persistent output accumulators (C layout per N-tile of i)

  for (int b = 0; b < RB; ++b) {
    // A fragments: A[z,i] = x[z,i] * shb[b,i]  (v_pk_mul_f16 on b128 LDS reads)
    V16 a1[3], a2[3];
#pragma unroll
    for (int ks = 0; ks < 3; ++ks)
#pragma unroll
      for (int g = 0; g < 2; ++g) {
        int kb = ks * 32 + hi * 16 + g * 8;
        v8h xa = *(const v8h*)&s_x1[wv][mrow * KP + kb];
        v8h xb = *(const v8h*)&s_x2[wv][mrow * KP + kb];
        v8h sb = *(const v8h*)&s_shb[b * KP + kb];
        a1[ks].h[g] = xa * sb;
        a2[ks].h[g] = xb * sb;
      }

    // forward: G1 = A1 * sha^T, G2 = A2 * sha^T; store G1.*G2 (f16) for transpose
#pragma unroll
    for (int nt = 0; nt < 4; ++nt) {
      v8f c1 = {}, c2 = {};
#pragma unroll
      for (int ks = 0; ks < 3; ++ks) {
        V16 bf;
        const _Float16* p = &s_Bf[((ks << 2) + nt) * 512 + (lane << 4)];
        bf.h[0] = *(const v8h*)p;
        bf.h[1] = *(const v8h*)(p + 8);
        c1 = __builtin_amdgcn_wmma_f32_16x16x32_f16(false, a1[ks].v, false, bf.v,
                                                    (short)0, c1, false, false);
        c2 = __builtin_amdgcn_wmma_f32_16x16x32_f16(false, a2[ks].v, false, bf.v,
                                                    (short)0, c2, false, false);
      }
#pragma unroll
      for (int r = 0; r < 8; ++r)
        s_g[wv][(r + 8 * hi) * AP + nt * 16 + mrow] = (_Float16)(c1[r] * c2[r]);
    }
    asm volatile("" ::: "memory");

    // reload G in A-fragment layout (C->A transpose via in-wave LDS round trip)
    V16 ag[2];
#pragma unroll
    for (int ks = 0; ks < 2; ++ks)
#pragma unroll
      for (int g = 0; g < 2; ++g) {
        int kb = ks * 32 + hi * 16 + g * 8;
        ag[ks].h[g] = *(const v8h*)&s_g[wv][mrow * AP + kb];
      }

    // backward: T = G * sha; acc += T * (shb[b,i]*qw[b])  (scale is per-lane scalar)
#pragma unroll
    for (int nt = 0; nt < 6; ++nt) {
      v8f c = {};
#pragma unroll
      for (int ks = 0; ks < 2; ++ks) {
        V16 bb;
        const _Float16* p = &s_Bb[(ks * 6 + nt) * 512 + (lane << 4)];
        bb.h[0] = *(const v8h*)p;
        bb.h[1] = *(const v8h*)(p + 8);
        c = __builtin_amdgcn_wmma_f32_16x16x32_f16(false, ag[ks].v, false, bb.v,
                                                   (short)0, c, false, false);
      }
      float w = s_shbw[b * KP + nt * 16 + mrow];
      acc[nt] += c * w;
    }
    asm volatile("" ::: "memory");
  }

  // ---- writeback: lane = column i, vgpr r = row z (C layout) ----
#pragma unroll
  for (int nt = 0; nt < 6; ++nt) {
    int i = nt * 16 + mrow;
    if (i < NDIM) {
#pragma unroll
      for (int r = 0; r < 8; ++r)
        out[(size_t)(z0 + r + 8 * hi) * NDIM + i] = acc[nt][r];
    }
  }
}

extern "C" void kernel_launch(void* const* d_in, const int* in_sizes, int n_in,
                              void* d_out, int out_size, void* d_ws, size_t ws_size,
                              hipStream_t stream) {
  const float* x1  = (const float*)d_in[0];
  const float* x2  = (const float*)d_in[1];
  const float* shb = (const float*)d_in[2];
  const float* sha = (const float*)d_in[3];
  const float* qw  = (const float*)d_in[4];
  float* out = (float*)d_out;
  int batch  = in_sizes[0] / NDIM;
  int tiles  = (batch + 15) / 16;
  int blocks = (tiles + WAVES - 1) / WAVES;
  gaunt_s2_wmma<<<blocks, TPB, 0, stream>>>(x1, x2, shb, sha, qw, out, batch);
}
